// Block_91018946937509
// MI455X (gfx1250) — matrix operand order repaired and checked by reading, not compiled, not run on hardware
//
#include <hip/hip_runtime.h>

// ---------------------------------------------------------------------------
// Fused WaveNet gated-residual layer for MI455X (gfx1250), bf16 WMMA.
//   z  [256 x T] = W1[256 x 352] @ Bpanel[352 x T] + zbias      (cond + dilated)
//   x  [128 x T] = tanh(z[0:128]) * sigmoid(z[128:256])
//   y  [384 x T] = W2[384 x 128] @ x + ybias   (rows 0..239 skip, 240..367 resid)
// Accumulators zero-init; biases folded into the epilogues so the WMMA K-loops
// carry no broadcast/copy chains.
// ---------------------------------------------------------------------------

#define BATCH   16
#define TLEN    8192
#define RCH     128
#define SCH     240
#define NMEL    80
#define DIL     8

#define T_TILE  64          // time columns per workgroup
#define KP      352         // padded K for stage-1 panel: 96(mel) + 128 + 128
#define M1      256         // stage-1 output rows (2R)
#define M2      384         // stage-2 padded output rows (240 skip + 128 res + 16 pad)

typedef __bf16 v8bf  __attribute__((ext_vector_type(8)));
typedef __bf16 v16bf __attribute__((ext_vector_type(16)));
typedef float  v8f   __attribute__((ext_vector_type(8)));
typedef unsigned short us8 __attribute__((ext_vector_type(8)));

__device__ __forceinline__ unsigned short f2bf(float f) {
    union { float f; unsigned int u; } v; v.f = f;
    unsigned int u = v.u;
    u += 0x7FFFu + ((u >> 16) & 1u);          // round-to-nearest-even
    return (unsigned short)(u >> 16);
}

__device__ __forceinline__ v16bf cat8(v8bf lo, v8bf hi) {
    return __builtin_shufflevector(lo, hi, 0,1,2,3,4,5,6,7,8,9,10,11,12,13,14,15);
}

// ---------------------------------------------------------------------------
// Prep: pack weights to bf16 in workspace + fuse biases. Tiny, runs once per
// launch, stream-ordered ahead of the main kernel.
// ---------------------------------------------------------------------------
__global__ void prep_kernel(const float* __restrict__ cond_w, const float* __restrict__ cond_b,
                            const float* __restrict__ dil_w,  const float* __restrict__ dil_b,
                            const float* __restrict__ skip_w, const float* __restrict__ skip_b,
                            const float* __restrict__ res_w,  const float* __restrict__ res_b,
                            unsigned short* __restrict__ w1, unsigned short* __restrict__ w2,
                            float* __restrict__ zbias, float* __restrict__ ybias)
{
    const int tid = blockIdx.x * blockDim.x + threadIdx.x;
    const int nth = gridDim.x * blockDim.x;

    // W1: [256 x KP]; K layout: [0,80)=cond_w, [80,96)=0, [96,224)=dil tap0, [224,352)=dil tap1
    for (int i = tid; i < M1 * KP; i += nth) {
        int o = i / KP, k = i - o * KP;
        float v = 0.f;
        if (k < NMEL)            v = cond_w[o * NMEL + k];
        else if (k >= 224)       v = dil_w[(o * RCH + (k - 224)) * 2 + 1];
        else if (k >= 96)        v = dil_w[(o * RCH + (k - 96)) * 2 + 0];
        w1[i] = f2bf(v);
    }
    // W2: [384 x 128]; rows [0,240)=skip_w, [240,368)=res_w, [368,384)=0
    for (int i = tid; i < M2 * RCH; i += nth) {
        int o = i >> 7, c = i & 127;
        float v = 0.f;
        if (o < SCH)             v = skip_w[o * RCH + c];
        else if (o < SCH + RCH)  v = res_w[(o - SCH) * RCH + c];
        w2[i] = f2bf(v);
    }
    for (int o = tid; o < M1; o += nth) zbias[o] = cond_b[o] + dil_b[o];
    for (int o = tid; o < M2; o += nth)
        ybias[o] = (o < SCH) ? skip_b[o] : ((o < SCH + RCH) ? res_b[o - SCH] : 0.f);
}

// ---------------------------------------------------------------------------
// Main fused kernel. 256 threads = 8 wave32.
// ---------------------------------------------------------------------------
__global__ __launch_bounds__(256) void wavenet_layer_kernel(
        const float* __restrict__ mel, const float* __restrict__ wav,
        const unsigned short* __restrict__ w1, const unsigned short* __restrict__ w2,
        const float* __restrict__ zbias, const float* __restrict__ ybias,
        float* __restrict__ out)
{
    // Activation panels, K-contiguous per time column so B-fragments are
    // two 16-byte LDS reads (at K and K+16).
    __shared__ unsigned short Blds[T_TILE * KP];    // 45056 B
    __shared__ unsigned short Xlds[T_TILE * RCH];   // 16384 B

    const int b    = blockIdx.y;
    const int t0   = blockIdx.x * T_TILE;
    const int tid  = threadIdx.x;
    const int lane = tid & 31;
    const int wv   = tid >> 5;          // wave id 0..7 (wave-uniform)
    const int l16  = lane & 15;
    const int hi   = lane >> 4;         // 0: lanes 0-15, 1: lanes 16-31

    // ---- stage 0: fill Blds (mel | wav(t-DIL) | wav(t)) as bf16 ------------
    for (int i = tid; i < 96 * T_TILE; i += 256) {      // t fastest -> coalesced
        int t = i & (T_TILE - 1), k = i >> 6;
        float v = (k < NMEL) ? mel[((long long)b * NMEL + k) * TLEN + t0 + t] : 0.f;
        Blds[t * KP + k] = f2bf(v);
    }
    for (int i = tid; i < RCH * T_TILE; i += 256) {
        int t = i & (T_TILE - 1), c = i >> 6;
        int gt = t0 + t;
        const float* wp = wav + ((long long)b * RCH + c) * TLEN;
        float cur  = wp[gt];
        float prev = (gt >= DIL) ? wp[gt - DIL] : 0.f;  // causal left pad
        Blds[t * KP +  96 + c] = f2bf(prev);
        Blds[t * KP + 224 + c] = f2bf(cur);
    }
    __syncthreads();

    // ---- stage 1: z = W1 @ Bpanel (bias folded into gate epilogue) --------
    // Wave wv: M-tile pair (rows 16wv.. : tanh half) and (rows 128+16wv.. : sigmoid half)
    const int mT = 16 * wv;
    const int mS = RCH + 16 * wv;

    v8f accT[4] = {}, accS[4] = {};     // zero-init: no broadcast copies in loop

    for (int kc = 0; kc < KP / 32; ++kc) {
        const int k0  = kc * 32;
        const int sel = hi * 8;         // A layout: lanes>=16 start at K+8
        const v8bf* paT = (const v8bf*)&w1[(mT + l16) * KP + k0 + sel];
        const v8bf* paS = (const v8bf*)&w1[(mS + l16) * KP + k0 + sel];
        v16bf aT = cat8(paT[0], paT[2]);    // K {0..7,16..23} (+8 if hi)
        v16bf aS = cat8(paS[0], paS[2]);
        #pragma unroll
        for (int nt = 0; nt < 4; ++nt) {
            const v8bf* pb = (const v8bf*)&Blds[(nt * 16 + l16) * KP + k0 + hi * 8];
            v16bf bf = cat8(pb[0], pb[2]);  // B layout: K {0..7,16..23} (+8 if hi)
            accT[nt] = __builtin_amdgcn_wmma_f32_16x16x32_bf16(
                           false, aT, false, bf, (short)0, accT[nt], false, false);
            accS[nt] = __builtin_amdgcn_wmma_f32_16x16x32_bf16(
                           false, aS, false, bf, (short)0, accS[nt], false, false);
        }
    }

    // ---- gate (+bias) + repack x into LDS (B layout for stage 2) ----------
    float zbT[8], zbS[8];
    #pragma unroll
    for (int i = 0; i < 8; ++i) {
        zbT[i] = zbias[mT + i + 8 * hi];
        zbS[i] = zbias[mS + i + 8 * hi];
    }
    #pragma unroll
    for (int nt = 0; nt < 4; ++nt) {
        us8 pk;
        #pragma unroll
        for (int i = 0; i < 8; ++i) {
            float xt = tanhf(accT[nt][i] + zbT[i]);
            float xs = 1.f / (1.f + __expf(-(accS[nt][i] + zbS[i])));
            pk[i] = f2bf(xt * xs);
        }
        int t  = nt * 16 + l16;          // time column (N)
        int c0 = 16 * wv + 8 * hi;       // x channel base for this fragment row group
        *(us8*)&Xlds[t * RCH + c0] = pk;
    }
    __syncthreads();

    // ---- stage 2: y = W2 @ x (bias folded into store epilogue) ------------
    // Wave wv: 3 M-tiles, rows 48wv .. 48wv+47 (boundaries 240/368 are tile-aligned)
    v8f y[12] = {};
    for (int kc = 0; kc < RCH / 32; ++kc) {
        const int k0 = kc * 32;
        v16bf a[3];
        #pragma unroll
        for (int j = 0; j < 3; ++j) {
            const v8bf* pa = (const v8bf*)&w2[(48 * wv + 16 * j + l16) * RCH + k0 + hi * 8];
            a[j] = cat8(pa[0], pa[2]);
        }
        #pragma unroll
        for (int nt = 0; nt < 4; ++nt) {
            const v8bf* pb = (const v8bf*)&Xlds[(nt * 16 + l16) * RCH + k0 + hi * 8];
            v16bf bf = cat8(pb[0], pb[2]);
            #pragma unroll
            for (int j = 0; j < 3; ++j)
                y[j * 4 + nt] = __builtin_amdgcn_wmma_f32_16x16x32_bf16(
                                    false, a[j], false, bf, (short)0, y[j * 4 + nt], false, false);
        }
    }

    // ---- store: skip (rows<240), resid (+wav, rows 240..367), drop padding
    const long long SKIP_ELEMS = (long long)BATCH * SCH * TLEN;
    #pragma unroll
    for (int j = 0; j < 3; ++j) {
        int mb = 48 * wv + 16 * j;      // wave-uniform; 240/368 boundaries tile-aligned
        if (mb >= SCH + RCH) continue;  // zero-padded rows
        float yb[8];
        #pragma unroll
        for (int i = 0; i < 8; ++i) yb[i] = ybias[mb + i + 8 * hi];
        #pragma unroll
        for (int nt = 0; nt < 4; ++nt) {
            int n = t0 + nt * 16 + l16;
            #pragma unroll
            for (int i = 0; i < 8; ++i) {
                int m = mb + i + 8 * hi;
                float v = y[j * 4 + nt][i] + yb[i];
                if (mb < SCH) {
                    out[((long long)b * SCH + m) * TLEN + n] = v;
                } else {
                    int c = m - SCH;
                    out[SKIP_ELEMS + ((long long)b * RCH + c) * TLEN + n]
                        = v + wav[((long long)b * RCH + c) * TLEN + n];
                }
            }
        }
    }
}

// ---------------------------------------------------------------------------
extern "C" void kernel_launch(void* const* d_in, const int* in_sizes, int n_in,
                              void* d_out, int out_size, void* d_ws, size_t ws_size,
                              hipStream_t stream) {
    const float* mel    = (const float*)d_in[0];
    const float* wav    = (const float*)d_in[1];
    const float* cond_w = (const float*)d_in[2];
    const float* cond_b = (const float*)d_in[3];
    const float* dil_w  = (const float*)d_in[4];
    const float* dil_b  = (const float*)d_in[5];
    const float* skip_w = (const float*)d_in[6];
    const float* skip_b = (const float*)d_in[7];
    const float* res_w  = (const float*)d_in[8];
    const float* res_b  = (const float*)d_in[9];

    // Workspace layout (total ~275 KB): W1 bf16 | W2 bf16 | zbias f32 | ybias f32
    unsigned short* w1 = (unsigned short*)d_ws;
    unsigned short* w2 = w1 + (size_t)M1 * KP;
    float* zbias = (float*)(w2 + (size_t)M2 * RCH);
    float* ybias = zbias + M1;

    prep_kernel<<<64, 256, 0, stream>>>(cond_w, cond_b, dil_w, dil_b,
                                        skip_w, skip_b, res_w, res_b,
                                        w1, w2, zbias, ybias);

    dim3 grid(TLEN / T_TILE, BATCH);    // (128, 16)
    wavenet_layer_kernel<<<grid, 256, 0, stream>>>(mel, wav, w1, w2, zbias, ybias,
                                                   (float*)d_out);
}
